// GroupedQueryAttention_86500641342014
// MI455X (gfx1250) — matrix operand order repaired and checked by reading, not compiled, not run on hardware
//
#include <hip/hip_runtime.h>

// ---------- types ----------
typedef __attribute__((ext_vector_type(16))) __bf16 v16bf;
typedef __attribute__((ext_vector_type(8)))  __bf16 v8bf;
typedef __attribute__((ext_vector_type(8)))  float  v8f;

static __device__ __forceinline__ v16bf cat8(v8bf lo, v8bf hi) {
    return __builtin_shufflevector(lo, hi, 0,1,2,3,4,5,6,7,8,9,10,11,12,13,14,15);
}
static __device__ __forceinline__ v8f wmma_bf16(v16bf a, v16bf b, v8f c) {
    // D(f32 16x16) = A(16x32 bf16) x B(32x16 bf16) + C
    return __builtin_amdgcn_wmma_f32_16x16x32_bf16(false, a, false, b, (short)0, c, false, false);
}

// ---------- problem constants ----------
// BSZ=2, SEQ=2048, DIM=2048, H=32, HD=64, KVH=8, REP=4. Tokens BS = 4096.

// ---------- f32 -> bf16 convert ----------
__global__ void k_cvt(const float* __restrict__ in, __bf16* __restrict__ out, long n) {
    long i = (long)blockIdx.x * blockDim.x + threadIdx.x;
    long stride = (long)gridDim.x * blockDim.x;
    for (; i < n; i += stride) out[i] = (__bf16)in[i];
}

// ---------- f32 [K,N] -> bf16 [N,K] (weight transpose) ----------
__global__ void k_transpose(const float* __restrict__ in, __bf16* __restrict__ out, int K, int N) {
    long n = (long)K * N;
    long i = (long)blockIdx.x * blockDim.x + threadIdx.x;
    long stride = (long)gridDim.x * blockDim.x;
    for (; i < n; i += stride) {
        int k = (int)(i / N), c = (int)(i % N);
        out[(size_t)c * K + k] = (__bf16)in[i];
    }
}

// ---------- bf16 GEMM, 64x64 tile per wave: C[M,N] = A[M,K] * Bt[N,K]^T ----------
// Register blocking: 4 A-frags x 4 B-frags -> 16 WMMAs per k-step (4x fragment reuse).
// mode 0: bf16 C row-major; mode 1: bf16 C transposed (C[n,m], ld=M); mode 2: f32 C row-major
__global__ __launch_bounds__(256)
void k_gemm_bf16(const __bf16* __restrict__ A, const __bf16* __restrict__ Bt,
                 __bf16* __restrict__ Cb, float* __restrict__ Cf,
                 int M, int N, int Kd, int mode)
{
    const int lane = threadIdx.x & 31;
    const int wave = (int)((blockIdx.x * blockDim.x + threadIdx.x) >> 5);
    const int tN = N >> 6;
    const int tiles = (M >> 6) * tN;
    if (wave >= tiles) return;                      // whole wave uniform -> EXEC stays all-1 for WMMA
    const int mT = (wave / tN) << 6;
    const int nT = (wave % tN) << 6;
    const int lm = lane & 15, lh = lane >> 4;

    const __bf16* arow[4];
    const __bf16* brow[4];
    #pragma unroll
    for (int i = 0; i < 4; ++i)                     // A frag: row=lane%16, K=lh*16+[0..16)
        arow[i] = A + (size_t)(mT + i * 16 + lm) * Kd + (lh << 4);
    #pragma unroll
    for (int j = 0; j < 4; ++j)                     // B frag: col=lane%16, K=lh*16+[0..16)
        brow[j] = Bt + (size_t)(nT + j * 16 + lm) * Kd + (lh << 4);

    v8f acc[4][4];
    const v8f vzero = {0.f,0.f,0.f,0.f,0.f,0.f,0.f,0.f};
    #pragma unroll
    for (int i = 0; i < 4; ++i)
        #pragma unroll
        for (int j = 0; j < 4; ++j) acc[i][j] = vzero;

    for (int k0 = 0; k0 < Kd; k0 += 32) {
        v16bf a[4], b[4];
        #pragma unroll
        for (int i = 0; i < 4; ++i)
            a[i] = *(const v16bf*)(arow[i] + k0);
        #pragma unroll
        for (int j = 0; j < 4; ++j)
            b[j] = *(const v16bf*)(brow[j] + k0);
        #pragma unroll
        for (int i = 0; i < 4; ++i)
            #pragma unroll
            for (int j = 0; j < 4; ++j)
                acc[i][j] = wmma_bf16(a[i], b[j], acc[i][j]);
    }

    #pragma unroll
    for (int i = 0; i < 4; ++i) {
        #pragma unroll
        for (int j = 0; j < 4; ++j) {
            #pragma unroll
            for (int r = 0; r < 8; ++r) {           // C: M = r + 8*(lane/16), N = lane%16
                int row = mT + i * 16 + r + (lh << 3);
                int col = nT + j * 16 + lm;
                if (mode == 2)      Cf[(size_t)row * N + col] = acc[i][j][r];
                else if (mode == 1) Cb[(size_t)col * M + row] = (__bf16)acc[i][j][r];
                else                Cb[(size_t)row * N + col] = (__bf16)acc[i][j][r];
            }
        }
    }
}

// ---------- flash-style GQA attention ----------
// Q : bf16 [4096, 2048]  (token-major, head h at cols h*64..)
// Kb: bf16 [4096, 512]   (token-major, kv head g at cols g*64..)
// Vt: bf16 [512, 4096]   (V transposed: row = g*64+d, col = b*2048+s)
// O : bf16 [4096, 2048]
__global__ __launch_bounds__(256)
void k_attn(const __bf16* __restrict__ Q, const __bf16* __restrict__ Kb,
            const __bf16* __restrict__ Vt, __bf16* __restrict__ O)
{
    __shared__ __align__(32) __bf16 lds[8 * 512];   // per-wave 16x32 bf16 P slab
    const int lane = threadIdx.x & 31;
    const int slot = (int)(threadIdx.x >> 5) * 512;
    const int wave = (int)((blockIdx.x * blockDim.x + threadIdx.x) >> 5);
    const int head = wave >> 7;                     // 0..63 = b*32 + h
    const int qt   = wave & 127;                    // q tile (16 rows)
    const int b = head >> 5;
    const int h = head & 31;
    const int g = h >> 2;                           // kv head = h / N_REP
    const int lm = lane & 15, lh = lane >> 4;

    const __bf16* qbase = Q  + ((size_t)(b * 2048 + qt * 16 + lm)) * 2048 + h * 64 + (lh << 4);
    const __bf16* kbase = Kb + (size_t)b * 2048 * 512 + (size_t)g * 64 + (lh << 4);
    const __bf16* vbase = Vt + (size_t)g * 64 * 4096 + (size_t)b * 2048 + (lh << 4);

    // Q fragments for the two K-steps over head_dim=64 (loaded once)
    v16bf a0 = *(const v16bf*)(qbase);
    v16bf a1 = *(const v16bf*)(qbase + 32);

    const float scale = 0.125f;                     // 1/sqrt(64)
    const v8f vzero = {0.f,0.f,0.f,0.f,0.f,0.f,0.f,0.f};
    float m8[8], l8[8];
    v8f acc[4];
    #pragma unroll
    for (int r = 0; r < 8; ++r) { m8[r] = -3.0e38f; l8[r] = 0.f; }
    #pragma unroll
    for (int t = 0; t < 4; ++t) acc[t] = vzero;

    for (int kt = 0; kt < 2048; kt += 32) {         // 32 keys per iteration
        const __bf16* kr = kbase + (size_t)(kt + lm) * 512;
        v16bf b00 = *(const v16bf*)(kr);                  // keys kt..+15, d 0..31
        v16bf b01 = *(const v16bf*)(kr + 32);             // keys kt..+15, d 32..63
        v16bf b10 = *(const v16bf*)(kr + 16 * 512);       // keys kt+16..+31
        v16bf b11 = *(const v16bf*)(kr + 16 * 512 + 32);

        v8f s0 = wmma_bf16(a0, b00, vzero); s0 = wmma_bf16(a1, b01, s0);
        v8f s1 = wmma_bf16(a0, b10, vzero); s1 = wmma_bf16(a1, b11, s1);

        #pragma unroll
        for (int r = 0; r < 8; ++r) {               // row = r + 8*lh; 16-lane reductions over key axis
            float v = fmaxf(s0[r], s1[r]);
            v = fmaxf(v, __shfl_xor(v, 1));
            v = fmaxf(v, __shfl_xor(v, 2));
            v = fmaxf(v, __shfl_xor(v, 4));
            v = fmaxf(v, __shfl_xor(v, 8));
            v *= scale;
            float nm = fmaxf(m8[r], v);
            float al = __expf(m8[r] - nm);
            float p0 = __expf(s0[r] * scale - nm);
            float p1 = __expf(s1[r] * scale - nm);
            float rs = p0 + p1;
            rs += __shfl_xor(rs, 1);
            rs += __shfl_xor(rs, 2);
            rs += __shfl_xor(rs, 4);
            rs += __shfl_xor(rs, 8);
            l8[r] = l8[r] * al + rs;
            m8[r] = nm;
            acc[0][r] *= al; acc[1][r] *= al; acc[2][r] *= al; acc[3][r] *= al;
            int row = r + (lh << 3);
            lds[slot + row * 32 + lm]      = (__bf16)p0;   // P -> row-major 16x32 bf16 in LDS
            lds[slot + row * 32 + 16 + lm] = (__bf16)p1;
        }
        asm volatile("s_wait_dscnt 0" ::: "memory");       // CDNA5 split counter: DS stores visible

        const __bf16* lp = &lds[slot + lm * 32 + (lh << 4)];
        v16bf aP = *(const v16bf*)lp;   // P as 16x32 A fragment

        #pragma unroll
        for (int t = 0; t < 4; ++t) {               // O += P @ V, 4 tiles of 16 d-cols
            v16bf bv = *(const v16bf*)(vbase + (size_t)(t * 16 + lm) * 4096 + kt);
            acc[t] = wmma_bf16(aP, bv, acc[t]);
        }
    }

    #pragma unroll
    for (int r = 0; r < 8; ++r) {
        float inv = 1.0f / l8[r];
        size_t orow = ((size_t)(b * 2048 + qt * 16 + r + (lh << 3))) * 2048 + h * 64 + lm;
        O[orow]      = (__bf16)(acc[0][r] * inv);
        O[orow + 16] = (__bf16)(acc[1][r] * inv);
        O[orow + 32] = (__bf16)(acc[2][r] * inv);
        O[orow + 48] = (__bf16)(acc[3][r] * inv);
    }
}

// ---------- host orchestration ----------
extern "C" void kernel_launch(void* const* d_in, const int* in_sizes, int n_in,
                              void* d_out, int out_size, void* d_ws, size_t ws_size,
                              hipStream_t stream) {
    (void)in_sizes; (void)n_in; (void)out_size; (void)ws_size;
    const float* x  = (const float*)d_in[0];
    const float* wq = (const float*)d_in[1];
    const float* wk = (const float*)d_in[2];
    const float* wv = (const float*)d_in[3];
    const float* wo = (const float*)d_in[4];
    float* out = (float*)d_out;

    char* ws = (char*)d_ws;
    size_t off = 0;
    auto take = [&](size_t bytes) -> void* {
        off = (off + 255) & ~(size_t)255;
        void* p = ws + off;
        off += bytes;
        return p;
    };
    __bf16* Xb  = (__bf16*)take(4096ull * 2048 * 2);   // x bf16
    __bf16* WqT = (__bf16*)take(2048ull * 2048 * 2);   // wq^T
    __bf16* WkT = (__bf16*)take(512ull  * 2048 * 2);   // wk^T
    __bf16* WvT = (__bf16*)take(512ull  * 2048 * 2);   // wv^T
    __bf16* WoT = (__bf16*)take(2048ull * 2048 * 2);   // wo^T
    __bf16* Qb  = (__bf16*)take(4096ull * 2048 * 2);   // Q
    __bf16* Kb  = (__bf16*)take(4096ull * 512  * 2);   // K
    __bf16* Vt  = (__bf16*)take(512ull  * 4096 * 2);   // V^T
    __bf16* At  = (__bf16*)take(4096ull * 2048 * 2);   // attention output

    k_cvt<<<2048, 256, 0, stream>>>(x, Xb, 4096l * 2048);
    k_transpose<<<2048, 256, 0, stream>>>(wq, WqT, 2048, 2048);
    k_transpose<<<512,  256, 0, stream>>>(wk, WkT, 2048, 512);
    k_transpose<<<512,  256, 0, stream>>>(wv, WvT, 2048, 512);
    k_transpose<<<2048, 256, 0, stream>>>(wo, WoT, 2048, 2048);

    // Q = X @ Wq   [4096,2048] : (4096/64)*(2048/64) = 2048 waves
    k_gemm_bf16<<<2048 / 8, 256, 0, stream>>>(Xb, WqT, Qb, nullptr, 4096, 2048, 2048, 0);
    // K = X @ Wk   [4096,512] : 64*8 = 512 waves
    k_gemm_bf16<<<512 / 8, 256, 0, stream>>>(Xb, WkT, Kb, nullptr, 4096, 512, 2048, 0);
    // V^T = (X @ Wv)^T  [512,4096]
    k_gemm_bf16<<<512 / 8, 256, 0, stream>>>(Xb, WvT, Vt, nullptr, 4096, 512, 2048, 1);

    // flash GQA attention: 64 heads x 128 q-tiles = 8192 waves, 8 waves/block
    k_attn<<<(64 * 128) / 8, 256, 0, stream>>>(Qb, Kb, Vt, At);

    // out = Attn @ Wo  -> f32 [4096,2048]
    k_gemm_bf16<<<2048 / 8, 256, 0, stream>>>(At, WoT, nullptr, out, 4096, 2048, 2048, 2);
}
